// Merge_Model_73701638799738
// MI455X (gfx1250) — compile-verified, matrix operand order checked by reading, not run on hardware
//
#include <hip/hip_runtime.h>
#include <hip/hip_bf16.h>
#include <math.h>

typedef __bf16 bf16_t;
typedef __attribute__((ext_vector_type(16))) __bf16 v16bf;
typedef __attribute__((ext_vector_type(8)))  float  v8f;

#define DV     300
#define N2DST  20000
#define E2N    300000
#define N1DST  8000
#define E1N    120000
#define BB     64
#define SS     512
#define CC     20
#define K2PAD  608     /* 2*D=600 padded to mult of 32 */
#define DPAD   320     /* D=300 padded to mult of 32   */
#define NSAGEP 320     /* N=300 padded to mult of 32   */
#define G4H    1200    /* 4*H                          */
#define G4HP   1216    /* 4*H padded to mult of 32     */

// ---------------------------------------------------------------------------
// bf16 WMMA GEMM with 2x2 register tiling: each wave computes a 32x32 block.
//   Cout[M,N] = A[M,Kpad] @ Bt[Npad,Kpad]^T (+bias0+bias1)(+Cadd)
// Fragment loads follow the CDNA5 ISA VGPR layouts:
//   A 16x32 bf16: lane m=L%16; vgpr v holds K = (v>=4?16:0)+(L>=16?8:0)+(v&3)*2 (+1)
//   B symmetric with N in place of M (B supplied pre-transposed, row = N).
//   C f32 16x16:  lane col=L%16; vgpr r holds row = r + 8*(L>=16).
// Per k-step: 2 A-frags + 2 B-frags (8 x b128) feed 4 WMMAs.
// ---------------------------------------------------------------------------
__global__ void __launch_bounds__(128)
wmma_gemm_bf16(const bf16_t* __restrict__ A, int lda,
               const int* __restrict__ rowmap,
               const bf16_t* __restrict__ Bt, int ldb,
               const float* __restrict__ bias0, const float* __restrict__ bias1,
               const float* __restrict__ Cadd, long cadd_rs,
               float* __restrict__ Cout, long ldc,
               int mtiles, int ntiles, int N, int Kpad)
{
    int wave = threadIdx.x >> 5;
    int lane = threadIdx.x & 31;
    int tile = blockIdx.x * 4 + wave;
    if (tile >= mtiles * ntiles) return;     // wave-uniform: EXEC stays full
    int mt = tile / ntiles, nt = tile % ntiles;
    int lrow = lane & 15;
    int khi  = (lane >> 4) << 3;             // 0 for lanes 0-15, 8 for 16-31

    int ar0 = mt * 32 + lrow, ar1 = ar0 + 16;
    if (rowmap) { ar0 = rowmap[ar0]; ar1 = rowmap[ar1]; }
    const bf16_t* ap0 = A  + (long)ar0 * lda;
    const bf16_t* ap1 = A  + (long)ar1 * lda;
    const bf16_t* bp0 = Bt + (long)(nt * 32 + lrow) * ldb;
    const bf16_t* bp1 = bp0 + (long)16 * ldb;

    v8f acc[2][2] = {};
    for (int k0 = 0; k0 < Kpad; k0 += 32) {
        v16bf a0, a1, b0, b1;
#pragma unroll
        for (int v = 0; v < 8; ++v) {
            int k = k0 + ((v & 4) << 2) + khi + ((v & 3) << 1);
            a0[2 * v] = ap0[k]; a0[2 * v + 1] = ap0[k + 1];
            a1[2 * v] = ap1[k]; a1[2 * v + 1] = ap1[k + 1];
            b0[2 * v] = bp0[k]; b0[2 * v + 1] = bp0[k + 1];
            b1[2 * v] = bp1[k]; b1[2 * v + 1] = bp1[k + 1];
        }
        acc[0][0] = __builtin_amdgcn_wmma_f32_16x16x32_bf16(
                        false, a0, false, b0, (short)0, acc[0][0], false, false);
        acc[0][1] = __builtin_amdgcn_wmma_f32_16x16x32_bf16(
                        false, a0, false, b1, (short)0, acc[0][1], false, false);
        acc[1][0] = __builtin_amdgcn_wmma_f32_16x16x32_bf16(
                        false, a1, false, b0, (short)0, acc[1][0], false, false);
        acc[1][1] = __builtin_amdgcn_wmma_f32_16x16x32_bf16(
                        false, a1, false, b1, (short)0, acc[1][1], false, false);
    }

#pragma unroll
    for (int f = 0; f < 2; ++f) {
        int col = nt * 32 + f * 16 + lrow;
        if (col < N) {
            float bias = 0.f;
            if (bias0) bias += bias0[col];
            if (bias1) bias += bias1[col];
#pragma unroll
            for (int e = 0; e < 2; ++e) {
#pragma unroll
                for (int r = 0; r < 8; ++r) {
                    int row = mt * 32 + e * 16 + r + khi;
                    float v = acc[e][f][r] + bias;
                    if (Cadd) v += Cadd[(long)row * cadd_rs + col];
                    Cout[(long)row * ldc + col] = v;
                }
            }
        }
    }
}

// --------------------------- small utility kernels -------------------------
__global__ void fill_zero_u32(unsigned* p, long n)
{
    long t = (long)blockIdx.x * blockDim.x + threadIdx.x;
    if (t < n) p[t] = 0u;
}

// per-edge scatter-add of source features (segment-sum numerator)
__global__ void edge_scatter(const int* __restrict__ esrc, const int* __restrict__ edst,
                             const int* __restrict__ nid,  const float* __restrict__ feat,
                             float* __restrict__ s, long E)
{
    long t = (long)blockIdx.x * blockDim.x + threadIdx.x;
    if (t >= E * DV) return;
    int e = (int)(t / DV), d = (int)(t % DV);
    int srcn = esrc[e];
    if (nid) srcn = nid[srcn];
    atomicAdd(&s[(long)edst[e] * DV + d], feat[(long)srcn * DV + d]);
}

__global__ void edge_degree(const int* __restrict__ edst, float* __restrict__ deg, int E)
{
    int e = blockIdx.x * blockDim.x + threadIdx.x;
    if (e < E) atomicAdd(&deg[edst[e]], 1.0f);
}

// build X = bf16([h_dst | s/max(deg,1) | 0-pad])  [n_dst, K2PAD]
__global__ void sage_concat(const int* __restrict__ dst_nid, const float* __restrict__ featd,
                            const float* __restrict__ s, const float* __restrict__ deg,
                            bf16_t* __restrict__ X, int n_dst)
{
    long t = (long)blockIdx.x * blockDim.x + threadIdx.x;
    if (t >= (long)n_dst * K2PAD) return;
    int n = (int)(t / K2PAD), k = (int)(t % K2PAD);
    float v = 0.f;
    if (k < DV) {
        int r = dst_nid ? dst_nid[n] : n;
        v = featd[(long)r * DV + k];
    } else if (k < 2 * DV) {
        v = s[(long)n * DV + (k - DV)] / fmaxf(deg[n], 1.0f);
    }
    X[t] = (bf16_t)v;
}

// w[K,N] fp32 -> Wt[Npad,Kpad] bf16 (transpose + zero pad)
__global__ void conv_w_t(const float* __restrict__ w, bf16_t* __restrict__ Wt,
                         int K, int N, int Kpad, int Npad)
{
    long t = (long)blockIdx.x * blockDim.x + threadIdx.x;
    if (t >= (long)Npad * Kpad) return;
    int n = (int)(t / Kpad), k = (int)(t % Kpad);
    Wt[t] = (bf16_t)((n < N && k < K) ? w[(long)k * N + n] : 0.f);
}

// fp32 [M,N] -> bf16 [M,Npad] (zero K-pad)
__global__ void conv_act(const float* __restrict__ x, bf16_t* __restrict__ y,
                         long M, int N, int Npad)
{
    long t = (long)blockIdx.x * blockDim.x + threadIdx.x;
    if (t >= M * Npad) return;
    long m = t / Npad; int j = (int)(t % Npad);
    y[t] = (bf16_t)((j < N) ? x[m * N + j] : 0.f);
}

// 3-view scaled-dot self-attention pool -> bf16 rows of the lookup table
__global__ void attn_pool_kernel(const float* __restrict__ h0, const float* __restrict__ h1,
                                 const float* __restrict__ h2, bf16_t* __restrict__ doc_bf)
{
    int i = blockIdx.x * blockDim.x + threadIdx.x;
    if (i >= N1DST) return;
    const float* x0 = h0 + (long)i * DV;
    const float* x1 = h1 + (long)i * DV;
    const float* x2 = h2 + (long)i * DV;
    float g00 = 0, g01 = 0, g02 = 0, g11 = 0, g12 = 0, g22 = 0;
    for (int d = 0; d < DV; ++d) {
        float a = x0[d], b = x1[d], c = x2[d];
        g00 += a * a; g01 += a * b; g02 += a * c;
        g11 += b * b; g12 += b * c; g22 += c * c;
    }
    float sc = rsqrtf((float)DV);
    float G[3][3] = {{g00 * sc, g01 * sc, g02 * sc},
                     {g01 * sc, g11 * sc, g12 * sc},
                     {g02 * sc, g12 * sc, g22 * sc}};
    float w[3] = {0.f, 0.f, 0.f};
    for (int r = 0; r < 3; ++r) {
        float m  = fmaxf(G[r][0], fmaxf(G[r][1], G[r][2]));
        float e0 = __expf(G[r][0] - m), e1 = __expf(G[r][1] - m), e2 = __expf(G[r][2] - m);
        float s  = e0 + e1 + e2;
        w[0] += e0 / s; w[1] += e1 / s; w[2] += e2 / s;
    }
    bf16_t* o = doc_bf + (long)i * DPAD;
    for (int d = 0; d < DV; ++d)
        o[d] = (bf16_t)(w[0] * x0[d] + w[1] * x1[d] + w[2] * x2[d]);
}

__device__ __forceinline__ float sigm(float x) { return 1.f / (1.f + __expf(-x)); }

// LSTM cell (torch gate order i,f,g,o); emits fp32 h to ys and bf16 h for next GEMM
__global__ void lstm_cell(const float* __restrict__ gates, float* __restrict__ cst,
                          float* __restrict__ ys_t, bf16_t* __restrict__ hb)
{
    int t = blockIdx.x * blockDim.x + threadIdx.x;
    if (t >= BB * DV) return;
    int b = t / DV, j = t % DV;
    const float* g = gates + (long)b * G4H;
    float i = sigm(g[j]);
    float f = sigm(g[DV + j]);
    float gg = tanhf(g[2 * DV + j]);
    float o = sigm(g[3 * DV + j]);
    float c = f * cst[t] + i * gg;
    cst[t] = c;
    float h = o * tanhf(c);
    ys_t[(long)b * (SS * DV) + j] = h;
    hb[(long)b * DPAD + j] = (bf16_t)h;
}

// gather last valid timestep + final FC [64,300]@[300,20]+b
__global__ void final_fc(const float* __restrict__ ys, const int* __restrict__ len,
                         const float* __restrict__ W, const float* __restrict__ bias,
                         float* __restrict__ out)
{
    int t = blockIdx.x * blockDim.x + threadIdx.x;
    if (t >= BB * CC) return;
    int b = t / CC, c = t % CC;
    int idx = len[b] - 1;
    idx = idx < 0 ? 0 : (idx > SS - 1 ? SS - 1 : idx);
    const float* x = ys + ((long)b * SS + idx) * DV;
    float acc = bias[c];
    for (int k = 0; k < DV; ++k) acc += x[k] * W[k * CC + c];
    out[t] = acc;
}

// ---------------------------------------------------------------------------
static inline void launch_gemm(hipStream_t s, const bf16_t* A, int lda, const int* rowmap,
                               const bf16_t* Bt, int ldb, const float* b0, const float* b1,
                               const float* Cadd, long crs, float* C, long ldc,
                               int M, int N, int Npad, int Kpad)
{
    int mtiles = M / 32, ntiles = Npad / 32;   // 32x32 supertiles
    int tiles = mtiles * ntiles;
    wmma_gemm_bf16<<<(tiles + 3) / 4, 128, 0, s>>>(A, lda, rowmap, Bt, ldb, b0, b1,
                                                   Cadd, crs, C, ldc,
                                                   mtiles, ntiles, N, Kpad);
}

#define L1D(kern, n, ...)                                                     \
    do {                                                                      \
        long _n = (long)(n);                                                  \
        kern<<<(unsigned)((_n + 255) / 256), 256, 0, stream>>>(__VA_ARGS__);  \
    } while (0)

#define ZERO(ptr, bytes) L1D(fill_zero_u32, (long)(bytes) / 4, (unsigned*)(ptr), (long)(bytes) / 4)

extern "C" void kernel_launch(void* const* d_in, const int* in_sizes, int n_in,
                              void* d_out, int out_size, void* d_ws, size_t ws_size,
                              hipStream_t stream)
{
    (void)in_sizes; (void)n_in; (void)out_size; (void)ws_size;

    const float* emb = (const float*)d_in[0];
    const int *src2[3], *dst2[3], *es2[3], *ed2[3], *es1[3], *ed1[3];
    const float *w2[3], *b2[3], *w1[3], *b1[3];
    for (int v = 0; v < 3; ++v) {
        int base = 1 + 4 * v;
        src2[v] = (const int*)d_in[base];     dst2[v] = (const int*)d_in[base + 1];
        es2[v]  = (const int*)d_in[base + 2]; ed2[v]  = (const int*)d_in[base + 3];
        es1[v]  = (const int*)d_in[13 + 2 * v];
        ed1[v]  = (const int*)d_in[14 + 2 * v];
        w2[v] = (const float*)d_in[19 + 2 * v]; b2[v] = (const float*)d_in[20 + 2 * v];
        w1[v] = (const float*)d_in[25 + 2 * v]; b1[v] = (const float*)d_in[26 + 2 * v];
    }
    const float* wih[2] = {(const float*)d_in[31], (const float*)d_in[35]};
    const float* whh[2] = {(const float*)d_in[32], (const float*)d_in[36]};
    const float* bih[2] = {(const float*)d_in[33], (const float*)d_in[37]};
    const float* bhh[2] = {(const float*)d_in[34], (const float*)d_in[38]};
    const float* fc_w = (const float*)d_in[39];
    const float* fc_b = (const float*)d_in[40];
    const int* x_batch = (const int*)d_in[41];
    const int* len_b   = (const int*)d_in[42];
    float* out = (float*)d_out;

    // ---- workspace bump allocator --------------------------------------
    char* ws = (char*)d_ws;
    size_t off = 0;
    auto alloc = [&](size_t bytes) -> void* {
        off = (off + 255) & ~(size_t)255;
        void* p = ws + off;
        off += bytes;
        return p;
    };
    float*  s2   = (float*)alloc((size_t)N2DST * DV * 4);
    float*  deg2 = (float*)alloc((size_t)N2DST * 4);
    bf16_t* X2   = (bf16_t*)alloc((size_t)N2DST * K2PAD * 2);
    float*  h2v[3]; for (int v = 0; v < 3; ++v) h2v[v] = (float*)alloc((size_t)N2DST * DV * 4);
    float*  s1   = (float*)alloc((size_t)N1DST * DV * 4);
    float*  deg1 = (float*)alloc((size_t)N1DST * 4);
    bf16_t* X1   = (bf16_t*)alloc((size_t)N1DST * K2PAD * 2);
    float*  h1v[3]; for (int v = 0; v < 3; ++v) h1v[v] = (float*)alloc((size_t)N1DST * DV * 4);
    bf16_t* doc_bf = (bf16_t*)alloc((size_t)(N1DST + 1) * DPAD * 2);
    bf16_t* Wsage[6]; for (int v = 0; v < 6; ++v) Wsage[v] = (bf16_t*)alloc((size_t)NSAGEP * K2PAD * 2);
    bf16_t* wih_t[2], *whh_t[2];
    for (int l = 0; l < 2; ++l) {
        wih_t[l] = (bf16_t*)alloc((size_t)G4HP * DPAD * 2);
        whh_t[l] = (bf16_t*)alloc((size_t)G4HP * DPAD * 2);
    }
    float*  xW    = (float*)alloc((size_t)BB * SS * G4H * 4);
    float*  ys1   = (float*)alloc((size_t)BB * SS * DV * 4);
    float*  ys2   = (float*)alloc((size_t)BB * SS * DV * 4);
    bf16_t* ys1bf = (bf16_t*)alloc((size_t)BB * SS * DPAD * 2);
    float*  gates = (float*)alloc((size_t)BB * G4H * 4);
    bf16_t* hb    = (bf16_t*)alloc((size_t)BB * DPAD * 2);
    float*  cbuf  = (float*)alloc((size_t)BB * DV * 4);

    // ---- weight conversions (transpose to [Npad][Kpad] bf16) -----------
    for (int v = 0; v < 3; ++v) {
        L1D(conv_w_t, (long)NSAGEP * K2PAD, w2[v], Wsage[v],     2 * DV, DV, K2PAD, NSAGEP);
        L1D(conv_w_t, (long)NSAGEP * K2PAD, w1[v], Wsage[3 + v], 2 * DV, DV, K2PAD, NSAGEP);
    }
    for (int l = 0; l < 2; ++l) {
        L1D(conv_w_t, (long)G4HP * DPAD, wih[l], wih_t[l], DV, G4H, DPAD, G4HP);
        L1D(conv_w_t, (long)G4HP * DPAD, whh[l], whh_t[l], DV, G4H, DPAD, G4HP);
    }

    // ---- layer-2 SAGE on the three views -------------------------------
    for (int v = 0; v < 3; ++v) {
        ZERO(s2, (size_t)N2DST * DV * 4);
        ZERO(deg2, (size_t)N2DST * 4);
        L1D(edge_scatter, (long)E2N * DV, es2[v], ed2[v], src2[v], emb, s2, (long)E2N);
        L1D(edge_degree, E2N, ed2[v], deg2, E2N);
        L1D(sage_concat, (long)N2DST * K2PAD, dst2[v], emb, s2, deg2, X2, N2DST);
        launch_gemm(stream, X2, K2PAD, nullptr, Wsage[v], K2PAD, b2[v], nullptr,
                    nullptr, 0, h2v[v], DV, N2DST, DV, NSAGEP, K2PAD);
    }
    // ---- layer-1 SAGE ---------------------------------------------------
    for (int v = 0; v < 3; ++v) {
        ZERO(s1, (size_t)N1DST * DV * 4);
        ZERO(deg1, (size_t)N1DST * 4);
        L1D(edge_scatter, (long)E1N * DV, es1[v], ed1[v], (const int*)nullptr, h2v[v], s1, (long)E1N);
        L1D(edge_degree, E1N, ed1[v], deg1, E1N);
        L1D(sage_concat, (long)N1DST * K2PAD, (const int*)nullptr, h2v[v], s1, deg1, X1, N1DST);
        launch_gemm(stream, X1, K2PAD, nullptr, Wsage[3 + v], K2PAD, b1[v], nullptr,
                    nullptr, 0, h1v[v], DV, N1DST, DV, NSAGEP, K2PAD);
    }

    // ---- attention pool -> bf16 lookup table (row N1DST stays zero) ----
    ZERO(doc_bf, (size_t)(N1DST + 1) * DPAD * 2);
    L1D(attn_pool_kernel, N1DST, h1v[0], h1v[1], h1v[2], doc_bf);

    // ---- 2-layer LSTM ---------------------------------------------------
    for (int l = 0; l < 2; ++l) {
        const bf16_t* Ain   = (l == 0) ? doc_bf : ys1bf;
        const int*    rmap  = (l == 0) ? x_batch : nullptr;
        float*        ys    = (l == 0) ? ys1 : ys2;
        if (l == 1)
            L1D(conv_act, (long)BB * SS * DPAD, ys1, ys1bf, (long)BB * SS, DV, DPAD);
        // input projection for the whole sequence (gather fused via rowmap)
        launch_gemm(stream, Ain, DPAD, rmap, wih_t[l], DPAD, bih[l], bhh[l],
                    nullptr, 0, xW, G4H, BB * SS, G4H, G4HP, DPAD);
        ZERO(hb, (size_t)BB * DPAD * 2);
        ZERO(cbuf, (size_t)BB * DV * 4);
        for (int t = 0; t < SS; ++t) {
            launch_gemm(stream, hb, DPAD, nullptr, whh_t[l], DPAD, nullptr, nullptr,
                        xW + (long)t * G4H, (long)SS * G4H, gates, G4H,
                        BB, G4H, G4HP, DPAD);
            L1D(lstm_cell, BB * DV, gates, cbuf, ys + (long)t * DV, hb);
        }
    }

    // ---- final gather + FC ---------------------------------------------
    L1D(final_fc, BB * CC, ys2, len_b, fc_w, fc_b, out);
}
